// Layer_86732569575513
// MI455X (gfx1250) — matrix operand order repaired and checked by reading, not compiled, not run on hardware
//
#include <hip/hip_runtime.h>
#include <math.h>

// ---------------- types ----------------
typedef __attribute__((ext_vector_type(16))) __bf16 v16bf;
typedef __attribute__((ext_vector_type(8)))  __bf16 v8bf;
typedef __attribute__((ext_vector_type(8)))  float  v8f;

#define N_ROWS 2048
#define BATCH  8
#define DIM    1024

// ---------------- workspace layout (bytes) ----------------
// wn  : bf16 [2048][1024]            =   4 MiB
// xn  : bf16 [2048][8][1024]         =  32 MiB
// xT  : bf16 [8][1024][2048]         =  32 MiB
// sc  : f32  [2048][8][2048]         = 128 MiB
// wts : bf16 [2048][8][2048]         =  64 MiB
#define OFF_WN 0ull
#define OFF_XN (OFF_WN + 2048ull*1024*2)
#define OFF_XT (OFF_XN + 2048ull*8*1024*2)
#define OFF_SC (OFF_XT + 8ull*1024*2048*2)
#define OFF_W  (OFF_SC + 2048ull*8*2048*4)

// ---------------- reductions (wave32, 256-thread blocks = 8 waves) ----------------
__device__ __forceinline__ float blockReduceSum(float v, float* smem) {
    int lane = threadIdx.x & 31;
    int w    = threadIdx.x >> 5;
    #pragma unroll
    for (int o = 16; o > 0; o >>= 1) v += __shfl_down(v, o, 32);
    if (lane == 0) smem[w] = v;
    __syncthreads();
    float r = (threadIdx.x < 8) ? smem[threadIdx.x] : 0.0f;
    if (w == 0) {
        #pragma unroll
        for (int o = 4; o > 0; o >>= 1) r += __shfl_down(r, o, 32);
        if (lane == 0) smem[0] = r;
    }
    __syncthreads();
    float out = smem[0];
    __syncthreads();
    return out;
}

__device__ __forceinline__ float blockReduceMax(float v, float* smem) {
    int lane = threadIdx.x & 31;
    int w    = threadIdx.x >> 5;
    #pragma unroll
    for (int o = 16; o > 0; o >>= 1) v = fmaxf(v, __shfl_down(v, o, 32));
    if (lane == 0) smem[w] = v;
    __syncthreads();
    float r = (threadIdx.x < 8) ? smem[threadIdx.x] : -3.402823466e38f;
    if (w == 0) {
        #pragma unroll
        for (int o = 4; o > 0; o >>= 1) r = fmaxf(r, __shfl_down(r, o, 32));
        if (lane == 0) smem[0] = r;
    }
    __syncthreads();
    float out = smem[0];
    __syncthreads();
    return out;
}

// ---------------- WMMA helpers ----------------
__device__ __forceinline__ v8f wmma_bf16(v16bf a, v16bf b, v8f c) {
    // D = A(16x32) * B(32x16) + C, f32 accumulate
    return __builtin_amdgcn_wmma_f32_16x16x32_bf16(false, a, false, b, (short)0, c, false, false);
}

// A fragment (16x32 bf16): lane m = lane&15, khalf = lane>>4.
// Element order: K = khalf*16 + {0..15}.
__device__ __forceinline__ v16bf load_frag_a(const __bf16* __restrict__ row, int khalf) {
    v8bf lo = *(const v8bf*)(row + khalf * 16);
    v8bf hi = *(const v8bf*)(row + khalf * 16 + 8);
    v16bf r;
    #pragma unroll
    for (int i = 0; i < 8; ++i) { r[i] = lo[i]; r[i + 8] = hi[i]; }
    return r;
}

// B fragment (32x16 bf16): lane n = lane&15 holds column n (a row of B^T),
// ksel = lane>>4 selects K = ksel*16 + {0..15} (32 contiguous bytes).
__device__ __forceinline__ v16bf load_frag_b(const __bf16* __restrict__ row, int ksel) {
    return *(const v16bf*)(row + ksel * 16);
}

// ---------------- kernel 1: row L2 normalize f32 -> bf16 (rows of 1024) ----------------
__global__ void l2norm_rows_kernel(const float* __restrict__ in, __bf16* __restrict__ out) {
    __shared__ float red[8];
    const size_t row = blockIdx.x;
    const float* p = in + row * DIM;
    float v[4]; float ss = 0.0f;
    #pragma unroll
    for (int i = 0; i < 4; ++i) { v[i] = p[threadIdx.x + 256 * i]; ss += v[i] * v[i]; }
    float tot = blockReduceSum(ss, red);
    float scale = 1.0f / fmaxf(sqrtf(tot), 1e-12f);
    __bf16* q = out + row * DIM;
    #pragma unroll
    for (int i = 0; i < 4; ++i) q[threadIdx.x + 256 * i] = (__bf16)(v[i] * scale);
}

// ---------------- kernel 2: tiled transpose x[m][b][d] -> xT[b][d][m] (bf16) ----------------
__global__ void transpose_x_kernel(const float* __restrict__ x, __bf16* __restrict__ xT) {
    __shared__ float tile[32][33];
    const int b  = blockIdx.z;
    const int m0 = blockIdx.x * 32;
    const int d0 = blockIdx.y * 32;
    const int tx = threadIdx.x;   // 0..31
    const int ty = threadIdx.y;   // 0..7
    #pragma unroll
    for (int r = 0; r < 32; r += 8) {
        size_t m = m0 + ty + r;
        tile[ty + r][tx] = x[(m * BATCH + b) * DIM + d0 + tx];
    }
    __syncthreads();
    #pragma unroll
    for (int r = 0; r < 32; r += 8) {
        size_t d = d0 + ty + r;
        xT[((size_t)b * DIM + d) * N_ROWS + m0 + tx] = (__bf16)tile[tx][ty + r];
    }
}

// ---------------- kernel 3: GEMM1  scores[I][b][J] = wn[I,:] . xn[J,b,:] ----------------
// per-wave 32x64 output tile (2 M-frags x 4 N-frags = 8 WMMAs per K=32 step)
__global__ void gemm1_scores_kernel(const __bf16* __restrict__ wn,
                                    const __bf16* __restrict__ xn,
                                    float* __restrict__ scores) {
    const int lane = threadIdx.x & 31;
    const int wave = threadIdx.x >> 5;
    const int I0 = blockIdx.x * 32;
    const int J0 = (blockIdx.y * 8 + wave) * 64;
    const int b  = blockIdx.z;

    const int mA = lane & 15, khalf = lane >> 4;
    const int nB = lane & 15, ksel  = lane >> 4;

    const __bf16* arow[2];
    arow[0] = wn + (size_t)(I0 + mA) * DIM;
    arow[1] = wn + (size_t)(I0 + 16 + mA) * DIM;
    const __bf16* brow[4];
    #pragma unroll
    for (int ni = 0; ni < 4; ++ni)
        brow[ni] = xn + ((size_t)(J0 + 16 * ni + nB) * BATCH + b) * DIM;

    v8f acc[2][4] = {};
    for (int kc = 0; kc < DIM; kc += 32) {
        // speculative prefetch ~4 K-steps ahead (global_prefetch_b8; dropped past row end)
        __builtin_prefetch(arow[0] + kc + 128, 0, 1);
        __builtin_prefetch(arow[1] + kc + 128, 0, 1);
        #pragma unroll
        for (int ni = 0; ni < 4; ++ni) __builtin_prefetch(brow[ni] + kc + 128, 0, 1);

        v16bf a[2], bb[4];
        #pragma unroll
        for (int mi = 0; mi < 2; ++mi) a[mi] = load_frag_a(arow[mi] + kc, khalf);
        #pragma unroll
        for (int ni = 0; ni < 4; ++ni) bb[ni] = load_frag_b(brow[ni] + kc, ksel);
        #pragma unroll
        for (int mi = 0; mi < 2; ++mi)
            #pragma unroll
            for (int ni = 0; ni < 4; ++ni)
                acc[mi][ni] = wmma_bf16(a[mi], bb[ni], acc[mi][ni]);
    }

    const int cn = lane & 15, mBase = (lane >> 4) * 8;
    #pragma unroll
    for (int mi = 0; mi < 2; ++mi) {
        #pragma unroll
        for (int r = 0; r < 8; ++r) {
            size_t o = ((size_t)(I0 + mi * 16 + mBase + r) * BATCH + b) * N_ROWS + (J0 + cn);
            scores[o]      = acc[mi][0][r];
            scores[o + 16] = acc[mi][1][r];
            scores[o + 32] = acc[mi][2][r];
            scores[o + 48] = acc[mi][3][r];
        }
    }
}

// ---------------- kernel 4: softmax over J (rows of 2048), f32 -> bf16 ----------------
__global__ void softmax_rows_kernel(const float* __restrict__ scores, __bf16* __restrict__ wts) {
    __shared__ float red[8];
    const size_t row = (size_t)blockIdx.x * BATCH + blockIdx.y;
    const float* p = scores + row * N_ROWS;
    __bf16* q = wts + row * N_ROWS;
    float v[8]; float mx = -3.402823466e38f;
    #pragma unroll
    for (int i = 0; i < 8; ++i) { v[i] = p[threadIdx.x + 256 * i]; mx = fmaxf(mx, v[i]); }
    mx = blockReduceMax(mx, red);
    float s = 0.0f;
    #pragma unroll
    for (int i = 0; i < 8; ++i) { v[i] = __expf(v[i] - mx); s += v[i]; }
    s = blockReduceSum(s, red);
    float inv = 1.0f / s;
    #pragma unroll
    for (int i = 0; i < 8; ++i) q[threadIdx.x + 256 * i] = (__bf16)(v[i] * inv);
}

// ---------------- kernel 5: GEMM2  out[I][b][d] = sum_J wts[I][b][J] * xT[b][d][J] ----------------
// per-wave 32x64 output tile, K = 2048 in steps of 32
__global__ void gemm2_out_kernel(const __bf16* __restrict__ wts,
                                 const __bf16* __restrict__ xT,
                                 float* __restrict__ out) {
    const int lane = threadIdx.x & 31;
    const int wave = threadIdx.x >> 5;
    const int I0 = blockIdx.x * 32;
    const int d0 = (blockIdx.y * 8 + wave) * 64;
    const int b  = blockIdx.z;

    const int mA = lane & 15, khalf = lane >> 4;
    const int nB = lane & 15, ksel  = lane >> 4;

    const __bf16* arow[2];
    arow[0] = wts + ((size_t)(I0 + mA) * BATCH + b) * N_ROWS;
    arow[1] = wts + ((size_t)(I0 + 16 + mA) * BATCH + b) * N_ROWS;
    const __bf16* brow[4];
    #pragma unroll
    for (int ni = 0; ni < 4; ++ni)
        brow[ni] = xT + ((size_t)b * DIM + d0 + 16 * ni + nB) * N_ROWS;

    v8f acc[2][4] = {};
    for (int kc = 0; kc < N_ROWS; kc += 32) {
        __builtin_prefetch(arow[0] + kc + 128, 0, 1);
        __builtin_prefetch(arow[1] + kc + 128, 0, 1);
        #pragma unroll
        for (int ni = 0; ni < 4; ++ni) __builtin_prefetch(brow[ni] + kc + 128, 0, 1);

        v16bf a[2], bb[4];
        #pragma unroll
        for (int mi = 0; mi < 2; ++mi) a[mi] = load_frag_a(arow[mi] + kc, khalf);
        #pragma unroll
        for (int ni = 0; ni < 4; ++ni) bb[ni] = load_frag_b(brow[ni] + kc, ksel);
        #pragma unroll
        for (int mi = 0; mi < 2; ++mi)
            #pragma unroll
            for (int ni = 0; ni < 4; ++ni)
                acc[mi][ni] = wmma_bf16(a[mi], bb[ni], acc[mi][ni]);
    }

    const int cn = lane & 15, mBase = (lane >> 4) * 8;
    #pragma unroll
    for (int mi = 0; mi < 2; ++mi) {
        #pragma unroll
        for (int r = 0; r < 8; ++r) {
            size_t o = ((size_t)(I0 + mi * 16 + mBase + r) * BATCH + b) * DIM + (d0 + cn);
            out[o]      = acc[mi][0][r];
            out[o + 16] = acc[mi][1][r];
            out[o + 32] = acc[mi][2][r];
            out[o + 48] = acc[mi][3][r];
        }
    }
}

// ---------------- kernel 6: y = out + x ; LayerNorm over d (in-place on out) ----------------
__global__ void residual_ln_kernel(float* __restrict__ out, const float* __restrict__ x,
                                   const float* __restrict__ gamma, const float* __restrict__ beta) {
    __shared__ float red[8];
    const size_t base = ((size_t)blockIdx.x * BATCH + blockIdx.y) * DIM;
    float y[4]; float s = 0.0f, s2 = 0.0f;
    #pragma unroll
    for (int i = 0; i < 4; ++i) {
        int d = threadIdx.x + 256 * i;
        float t = out[base + d] + x[base + d];
        y[i] = t; s += t; s2 += t * t;
    }
    s  = blockReduceSum(s, red);
    s2 = blockReduceSum(s2, red);
    float mu  = s * (1.0f / DIM);
    float var = s2 * (1.0f / DIM) - mu * mu;
    float rr  = rsqrtf(var + 1e-5f);
    #pragma unroll
    for (int i = 0; i < 4; ++i) {
        int d = threadIdx.x + 256 * i;
        out[base + d] = (y[i] - mu) * rr * gamma[d] + beta[d];
    }
}

// ---------------- launcher ----------------
extern "C" void kernel_launch(void* const* d_in, const int* in_sizes, int n_in,
                              void* d_out, int out_size, void* d_ws, size_t ws_size,
                              hipStream_t stream) {
    const float* x     = (const float*)d_in[0];   // [2048][8][1024]
    const float* embed = (const float*)d_in[1];   // [2048][1024]
    const float* gamma = (const float*)d_in[2];   // [1024]
    const float* beta  = (const float*)d_in[3];   // [1024]
    float* out = (float*)d_out;                   // [2048][8][1024]

    char* ws = (char*)d_ws;
    __bf16* wn  = (__bf16*)(ws + OFF_WN);
    __bf16* xn  = (__bf16*)(ws + OFF_XN);
    __bf16* xT  = (__bf16*)(ws + OFF_XT);
    float*  sc  = (float*) (ws + OFF_SC);
    __bf16* wts = (__bf16*)(ws + OFF_W);

    // 1) normalize embed rows (2048) and x rows (2048*8) -> bf16
    l2norm_rows_kernel<<<N_ROWS, 256, 0, stream>>>(embed, wn);
    l2norm_rows_kernel<<<N_ROWS * BATCH, 256, 0, stream>>>(x, xn);

    // 2) transpose raw x per batch -> xT bf16 [b][d][m]
    transpose_x_kernel<<<dim3(N_ROWS / 32, DIM / 32, BATCH), dim3(32, 8), 0, stream>>>(x, xT);

    // 3) GEMM1: scores (8 waves/block, wave = one 32x64 tile)
    gemm1_scores_kernel<<<dim3(N_ROWS / 32, (N_ROWS / 64) / 8, BATCH), 256, 0, stream>>>(wn, xn, sc);

    // 4) softmax rows -> bf16 weights
    softmax_rows_kernel<<<dim3(N_ROWS, BATCH), 256, 0, stream>>>(sc, wts);

    // 5) GEMM2: out = weights . x   (via xT)
    gemm2_out_kernel<<<dim3(N_ROWS / 32, (DIM / 64) / 8, BATCH), 256, 0, stream>>>(wts, xT, out);

    // 6) residual + LayerNorm in-place
    residual_ln_kernel<<<dim3(N_ROWS, BATCH), 256, 0, stream>>>(out, x, gamma, beta);
}
